// _PointnetSAModuleBase_52896817217688
// MI455X (gfx1250) — matrix operand order repaired and checked by reading, not compiled, not run on hardware
//
#include <hip/hip_runtime.h>
#include <hip/hip_bf16.h>

// ---------------------------------------------------------------------------
// PointNet++ SA module for MI455X (gfx1250, wave32)
//   Phase 1: farthest point sampling    -> new_xyz   (d_out[0 : 4*1024*3))
//   Phase 2: ball query (first-64 hits) -> idx       (workspace, int32)
//   Phase 3: group + 3-layer MLP + max  -> features  (d_out[12288 : ...))
//            via v_wmma_f32_16x16x32_f16
// ---------------------------------------------------------------------------

#define BATCH   4
#define NPTS    16384
#define CFEAT   64
#define NPOINT  1024
#define NSAMPLE 64
#define R2      0.64f   // RADIUS^2, RADIUS = 0.8

typedef __attribute__((ext_vector_type(16))) _Float16 v16h;
typedef __attribute__((ext_vector_type(8)))  float    v8f;

// ---------------------------------------------------------------------------
// Phase 1: farthest point sampling. One 1024-thread block per batch.
// Each thread caches 16 points + running min-dists in registers; per
// iteration do min-update then a shfl+LDS argmax tree (first-index ties).
// ---------------------------------------------------------------------------
__global__ void __launch_bounds__(1024) fps_kernel(
    const float* __restrict__ xyz, float* __restrict__ new_xyz)
{
    const int b = blockIdx.x;
    const int t = threadIdx.x;
    const float* X = xyz + (size_t)b * NPTS * 3;

    float px[16], py[16], pz[16], dd[16];
#pragma unroll
    for (int i = 0; i < 16; ++i) {
        int p = t + i * 1024;
        px[i] = X[p * 3 + 0];
        py[i] = X[p * 3 + 1];
        pz[i] = X[p * 3 + 2];
        dd[i] = 1.0e10f;
    }

    __shared__ float s_val[32];
    __shared__ int   s_idx[32];
    __shared__ float s_pt[3];
    __shared__ int   s_cur;

    int cur = 0;
    for (int j = 0; j < NPOINT; ++j) {
        if (t == 0) {
            float cx = X[cur * 3 + 0], cy = X[cur * 3 + 1], cz = X[cur * 3 + 2];
            s_pt[0] = cx; s_pt[1] = cy; s_pt[2] = cz;
            // reference scan emits the carried index -> emit point `cur` now
            new_xyz[((size_t)b * NPOINT + j) * 3 + 0] = cx;
            new_xyz[((size_t)b * NPOINT + j) * 3 + 1] = cy;
            new_xyz[((size_t)b * NPOINT + j) * 3 + 2] = cz;
        }
        __syncthreads();
        const float cx = s_pt[0], cy = s_pt[1], cz = s_pt[2];

        float bestv = -1.0f;
        int   besti = 0;
#pragma unroll
        for (int i = 0; i < 16; ++i) {
            float dx = px[i] - cx, dy = py[i] - cy, dz = pz[i] - cz;
            float d  = dx * dx + dy * dy + dz * dz;
            dd[i] = fminf(dd[i], d);
            if (dd[i] > bestv) { bestv = dd[i]; besti = t + i * 1024; }
        }
        // wave32 argmax reduction (prefer lower index on ties)
#pragma unroll
        for (int off = 16; off > 0; off >>= 1) {
            float ov = __shfl_down(bestv, off);
            int   oi = __shfl_down(besti, off);
            if (ov > bestv || (ov == bestv && oi < besti)) { bestv = ov; besti = oi; }
        }
        if ((t & 31) == 0) { s_val[t >> 5] = bestv; s_idx[t >> 5] = besti; }
        __syncthreads();
        if (t < 32) {
            float v = s_val[t];
            int   ix = s_idx[t];
#pragma unroll
            for (int off = 16; off > 0; off >>= 1) {
                float ov = __shfl_down(v, off);
                int   oi = __shfl_down(ix, off);
                if (ov > v || (ov == v && oi < ix)) { v = ov; ix = oi; }
            }
            if (t == 0) s_cur = ix;
        }
        __syncthreads();
        cur = s_cur;
    }
}

// ---------------------------------------------------------------------------
// Phase 2: ball query. One wave per query; ballot/popc compaction collects
// the first 64 in-radius indices in ascending order, padded with the first.
// ---------------------------------------------------------------------------
__global__ void __launch_bounds__(256) ball_query_kernel(
    const float* __restrict__ xyz, const float* __restrict__ new_xyz,
    int* __restrict__ idx_out)
{
    const int lane = threadIdx.x & 31;
    const int wv   = threadIdx.x >> 5;
    const int q    = blockIdx.x * 8 + wv;     // 0..4095
    const int b    = q >> 10;

    const float qx = new_xyz[q * 3 + 0];
    const float qy = new_xyz[q * 3 + 1];
    const float qz = new_xyz[q * 3 + 2];
    const float* X = xyz + (size_t)b * NPTS * 3;
    int* out = idx_out + (size_t)q * NSAMPLE;

    int count = 0;
    int firstIdx = NPTS - 1;                  // no-hit case: min(N, N-1)

    for (int base = 0; base < NPTS; base += 32) {
        int p = base + lane;
        float dx = X[p * 3 + 0] - qx;
        float dy = X[p * 3 + 1] - qy;
        float dz = X[p * 3 + 2] - qz;
        bool pred = (dx * dx + dy * dy + dz * dz) < R2;
        unsigned mask = (unsigned)__ballot(pred);
        if (mask) {
            if (count == 0) firstIdx = __shfl(p, __ffs(mask) - 1);
            int pos = count + __popc(mask & ((1u << lane) - 1u));
            if (pred && pos < NSAMPLE) out[pos] = p;
            count += __popc(mask);
            if (count >= NSAMPLE) break;
        }
    }
    for (int pos = lane; pos < NSAMPLE; pos += 32)
        if (pos >= count) out[pos] = firstIdx;
}

// ---------------------------------------------------------------------------
// Phase 3: group + MLP(67->64->64->128) + max over 64 samples, via WMMA.
// 128 threads (4 waves) per query; wave w owns rows [16w,16w+16).
// ---------------------------------------------------------------------------

// A fragment: 16x32 f16 tile from LDS (row-major, given stride in halves).
// Layout: all lanes rows row0..row0+15 (lane&15); lanes<16 K {k0..k0+7,
// k0+16..k0+23}, lanes>=16 K {k0+8..k0+15, k0+24..k0+31}.
__device__ inline v16h load_A(const _Float16* lds, int stride, int row0, int k0, int lane)
{
    const _Float16* base = lds + (row0 + (lane & 15)) * stride;
    const int kb = k0 + ((lane & 16) ? 8 : 0);
    v16h a;
#pragma unroll
    for (int j = 0; j < 8; ++j) a[j] = base[kb + j];
#pragma unroll
    for (int j = 0; j < 8; ++j) a[8 + j] = base[kb + 16 + j];
    return a;
}

// B fragment: 32x16 f16 tile of W^T. Weights stored [out][in] row-major in
// LDS; element (k,n) = W[n0+n][k0+k]. lanes<16: K {k0..k0+7, k0+16..k0+23}
// (col n0+lane), lanes>=16: K {k0+8..k0+15, k0+24..k0+31}.
__device__ inline v16h load_B(const _Float16* w, int stride, int k0, int n0, int lane)
{
    const _Float16* base = w + (n0 + (lane & 15)) * stride + k0 + ((lane & 16) ? 8 : 0);
    v16h bfrag;
#pragma unroll
    for (int j = 0; j < 16; ++j) bfrag[j] = base[j + (j & 8)];
    return bfrag;
}

__device__ inline v8f wmma16(v16h a, v16h b, v8f c)
{
    return __builtin_amdgcn_wmma_f32_16x16x32_f16(
        /*neg_a=*/false, a, /*neg_b=*/false, b,
        /*c_mod=*/(short)0, c, /*reuse_a=*/false, /*reuse_b=*/false);
}

__global__ void __launch_bounds__(128) group_mlp_kernel(
    const float* __restrict__ xyz, const float* __restrict__ features,
    const float* __restrict__ W1, const float* __restrict__ b1,
    const float* __restrict__ W2, const float* __restrict__ b2,
    const float* __restrict__ W3, const float* __restrict__ b3,
    const float* __restrict__ new_xyz, const int* __restrict__ idx,
    float* __restrict__ out_feat)
{
    __shared__ _Float16 hs [64 * 96];    // h,  rows k=0..63, cols padded to 96
    __shared__ _Float16 xs [64 * 64];    // inter-layer activations
    __shared__ _Float16 w1s[64 * 96];    // W1  [64][96]  (cols>=67 zero)
    __shared__ _Float16 w2s[64 * 64];    // W2  [64][64]
    __shared__ _Float16 w3s[128 * 64];   // W3  [128][64]
    __shared__ float    red[4][128];

    const int tid  = threadIdx.x;
    const int lane = tid & 31;
    const int wv   = tid >> 5;
    const int q    = blockIdx.x;         // 0..4095
    const int b    = q >> 10;
    const int m    = q & 1023;

    // ---- stage weights (f32 -> f16, W1 zero-padded to stride 96) ----
    for (int i = tid; i < 64 * 96; i += 128) {
        int o = i / 96, c = i % 96;
        w1s[i] = (c < 67) ? (_Float16)W1[o * 67 + c] : (_Float16)0.0f;
    }
    for (int i = tid; i < 64 * 64; i += 128)  w2s[i] = (_Float16)W2[i];
    for (int i = tid; i < 128 * 64; i += 128) w3s[i] = (_Float16)W3[i];

    // ---- stage h = [g_xyz - q | g_feat], 64 x 96 (pad zero) ----
    const float qx = new_xyz[q * 3 + 0];
    const float qy = new_xyz[q * 3 + 1];
    const float qz = new_xyz[q * 3 + 2];
    const int* gidx = idx + (size_t)q * NSAMPLE;
    for (int i = tid; i < 64 * 96; i += 128) {
        int k = i / 96, c = i % 96;
        int p = gidx[k];
        float v = 0.0f;
        if      (c == 0) v = xyz[((size_t)b * NPTS + p) * 3 + 0] - qx;
        else if (c == 1) v = xyz[((size_t)b * NPTS + p) * 3 + 1] - qy;
        else if (c == 2) v = xyz[((size_t)b * NPTS + p) * 3 + 2] - qz;
        else if (c < 67) v = features[((size_t)b * CFEAT + (c - 3)) * NPTS + p];
        hs[i] = (_Float16)v;
    }
    __syncthreads();

    const int r0 = wv * 16;                     // row tile owned by this wave
    const int rbase = r0 + ((lane & 16) ? 8 : 0);
    const int cl = lane & 15;

    // ---- layer 1: hs(64x96) @ W1^T -> xs(64x64), ReLU ----
    {
        v16h a0 = load_A(hs, 96, r0,  0, lane);
        v16h a1 = load_A(hs, 96, r0, 32, lane);
        v16h a2 = load_A(hs, 96, r0, 64, lane);
#pragma unroll
        for (int ot = 0; ot < 4; ++ot) {
            v8f acc = {};
            acc = wmma16(a0, load_B(w1s, 96,  0, ot * 16, lane), acc);
            acc = wmma16(a1, load_B(w1s, 96, 32, ot * 16, lane), acc);
            acc = wmma16(a2, load_B(w1s, 96, 64, ot * 16, lane), acc);
            int   c    = ot * 16 + cl;
            float bias = b1[c];
#pragma unroll
            for (int i = 0; i < 8; ++i)
                xs[(rbase + i) * 64 + c] = (_Float16)fmaxf(acc[i] + bias, 0.0f);
        }
    }

    // ---- layer 2: xs(64x64) @ W2^T -> xs(64x64), ReLU (A prefetched) ----
    {
        v16h a0 = load_A(xs, 64, r0,  0, lane);
        v16h a1 = load_A(xs, 64, r0, 32, lane);
#pragma unroll
        for (int ot = 0; ot < 4; ++ot) {
            v8f acc = {};
            acc = wmma16(a0, load_B(w2s, 64,  0, ot * 16, lane), acc);
            acc = wmma16(a1, load_B(w2s, 64, 32, ot * 16, lane), acc);
            int   c    = ot * 16 + cl;
            float bias = b2[c];
#pragma unroll
            for (int i = 0; i < 8; ++i)
                xs[(rbase + i) * 64 + c] = (_Float16)fmaxf(acc[i] + bias, 0.0f);
        }
    }

    // ---- layer 3: xs(64x64) @ W3^T (128 out), in-register row max ----
    {
        v16h a0 = load_A(xs, 64, r0,  0, lane);
        v16h a1 = load_A(xs, 64, r0, 32, lane);
#pragma unroll
        for (int ot = 0; ot < 8; ++ot) {
            v8f acc = {};
            acc = wmma16(a0, load_B(w3s, 64,  0, ot * 16, lane), acc);
            acc = wmma16(a1, load_B(w3s, 64, 32, ot * 16, lane), acc);
            float mv = acc[0];
#pragma unroll
            for (int i = 1; i < 8; ++i) mv = fmaxf(mv, acc[i]);
            mv = fmaxf(mv, __shfl_xor(mv, 16));      // merge both row halves
            if (lane < 16) red[wv][ot * 16 + cl] = mv;
        }
    }
    __syncthreads();

    // ---- cross-wave max + bias + ReLU (relu(max+b) == max(relu(.+b))) ----
    {
        int c = tid;                                   // 0..127
        float v = fmaxf(fmaxf(red[0][c], red[1][c]), fmaxf(red[2][c], red[3][c]));
        v = fmaxf(v + b3[c], 0.0f);
        out_feat[((size_t)b * 128 + c) * NPOINT + m] = v;
    }
}

// ---------------------------------------------------------------------------
extern "C" void kernel_launch(void* const* d_in, const int* in_sizes, int n_in,
                              void* d_out, int out_size, void* d_ws, size_t ws_size,
                              hipStream_t stream)
{
    const float* xyz      = (const float*)d_in[0];
    const float* features = (const float*)d_in[1];
    const float* W1       = (const float*)d_in[2];
    const float* b1       = (const float*)d_in[3];
    const float* W2       = (const float*)d_in[4];
    const float* b2       = (const float*)d_in[5];
    const float* W3       = (const float*)d_in[6];
    const float* b3       = (const float*)d_in[7];

    float* new_xyz  = (float*)d_out;                       // (4,1024,3)
    float* out_feat = (float*)d_out + BATCH * NPOINT * 3;  // (4,128,1024)
    int*   ball_idx = (int*)d_ws;                          // (4,1024,64) int32

    fps_kernel<<<BATCH, 1024, 0, stream>>>(xyz, new_xyz);
    ball_query_kernel<<<(BATCH * NPOINT) / 8, 256, 0, stream>>>(xyz, new_xyz, ball_idx);
    group_mlp_kernel<<<BATCH * NPOINT, 128, 0, stream>>>(
        xyz, features, W1, b1, W2, b2, W3, b3, new_xyz, ball_idx, out_feat);
}
